// SymbioticBasisRefinement_18769007083985
// MI455X (gfx1250) — matrix operand order repaired and checked by reading, not compiled, not run on hardware
//
#include <hip/hip_runtime.h>
#include <hip/hip_bf16.h>

typedef __bf16 bf16;
typedef __attribute__((ext_vector_type(16))) __bf16 v16bf;
typedef __attribute__((ext_vector_type(8)))  float  v8f;

#define DIM 1024
#define NATOMS 64
#define NROWS 16384              // B*S = 4*4096
#define ROWS_PER_BLK 64          // 2 waves x 32 rows
#define NBLK (NROWS / ROWS_PER_BLK)   // 256

static __device__ __forceinline__ v8f wmma_bf16(v16bf a, v16bf b, v8f c) {
  // (neg_a, A, neg_b, B, c_mod, C, reuse_a, reuse_b)
  return __builtin_amdgcn_wmma_f32_16x16x32_bf16(false, a, false, b, (short)0, c,
                                                 false, false);
}

// ---------------------------------------------------------------------------
// Phase A1: K[a,e] = sum_f basis[a,f] * Wk[e,f] + bk[e]     (64 x 1024, f32)
// One wave per 16x16 output tile. grid = (4 atom-tiles, 64 e-tiles).
// ---------------------------------------------------------------------------
__global__ __launch_bounds__(32) void kernel_K(const float* __restrict__ basis,
                                               const float* __restrict__ Wk,
                                               const float* __restrict__ bk,
                                               float* __restrict__ Kmat) {
  const int lane = threadIdx.x;
  const int m = lane & 15, half = lane >> 4;
  const int a0 = blockIdx.x * 16;
  const int e0 = blockIdx.y * 16;
  const float* brow = basis + (a0 + m) * DIM;   // A operand: atom rows
  const float* wrow = Wk + (e0 + m) * DIM;      // B operand: column e0+m = Wk row
  const int off = half * 8;
  v8f acc = {};
  for (int f0 = 0; f0 < DIM; f0 += 32) {
    v16bf A, B;
    const float* ap = brow + f0 + half * 16;
#pragma unroll
    for (int i = 0; i < 16; ++i) A[i] = (bf16)ap[i];
    const float* bp = wrow + f0 + half * 16;
#pragma unroll
    for (int i = 0; i < 16; ++i) B[i] = (bf16)bp[i];
    acc = wmma_bf16(A, B, acc);
  }
  const float bkv = bk[e0 + m];
#pragma unroll
  for (int r = 0; r < 8; ++r)
    Kmat[(a0 + r + 8 * half) * DIM + e0 + m] = acc[r] + bkv;
}

// ---------------------------------------------------------------------------
// Phase A2: Mt[a][d] = scale * sum_e K[a,e] * Wq[e,d]   (stored bf16, [64][1024])
// ---------------------------------------------------------------------------
__global__ __launch_bounds__(256) void kernel_Mt(const float* __restrict__ Kmat,
                                                 const float* __restrict__ Wq,
                                                 bf16* __restrict__ Mt,
                                                 float scale) {
  const int a0 = (blockIdx.x >> 2) * 4;
  const int d  = (blockIdx.x & 3) * 256 + threadIdx.x;
  float acc0 = 0.f, acc1 = 0.f, acc2 = 0.f, acc3 = 0.f;
  for (int e = 0; e < DIM; ++e) {
    const float w = Wq[e * DIM + d];
    acc0 += Kmat[(a0 + 0) * DIM + e] * w;
    acc1 += Kmat[(a0 + 1) * DIM + e] * w;
    acc2 += Kmat[(a0 + 2) * DIM + e] * w;
    acc3 += Kmat[(a0 + 3) * DIM + e] * w;
  }
  Mt[(a0 + 0) * DIM + d] = (bf16)(acc0 * scale);
  Mt[(a0 + 1) * DIM + d] = (bf16)(acc1 * scale);
  Mt[(a0 + 2) * DIM + d] = (bf16)(acc2 * scale);
  Mt[(a0 + 3) * DIM + d] = (bf16)(acc3 * scale);
}

// ---------------------------------------------------------------------------
// Phase A3: c[a] = scale * sum_e bq[e] * K[a,e]
// ---------------------------------------------------------------------------
__global__ __launch_bounds__(256) void kernel_c(const float* __restrict__ Kmat,
                                                const float* __restrict__ bq,
                                                float* __restrict__ cvec,
                                                float scale) {
  __shared__ float red[256];
  const int a = blockIdx.x, t = threadIdx.x;
  float s = 0.f;
  for (int e = t; e < DIM; e += 256) s += bq[e] * Kmat[a * DIM + e];
  red[t] = s;
  __syncthreads();
  for (int st = 128; st > 0; st >>= 1) {
    if (t < st) red[t] += red[t + st];
    __syncthreads();
  }
  if (t == 0) cvec[a] = red[0] * scale;
}

// ---------------------------------------------------------------------------
// Phase A4: basisT[d][a] = bf16(basis[a][d])   ([1024][64], coalesced writes)
// ---------------------------------------------------------------------------
__global__ __launch_bounds__(256) void kernel_bt(const float* __restrict__ basis,
                                                 bf16* __restrict__ basisT) {
  const int idx = blockIdx.x * 256 + threadIdx.x;  // idx = d*64 + a
  const int d = idx >> 6, a = idx & 63;
  basisT[idx] = (bf16)basis[a * DIM + d];
}

// ---------------------------------------------------------------------------
// Phase B (main): 64 rows/block, 2 waves, 32 rows/wave (2 A-tiles share B):
//   attn = x @ Mt^T + c  (bf16 WMMA, f32 acc) -> softmax + entropy (LDS)
//   x_clean = weights @ basis (bf16 WMMA) -> coalesced f32 stores
// ---------------------------------------------------------------------------
__global__ __launch_bounds__(64) void kernel_main(const float* __restrict__ x,
                                                  const bf16* __restrict__ Mt,
                                                  const float* __restrict__ cvec,
                                                  const bf16* __restrict__ basisT,
                                                  float* __restrict__ out,
                                                  float* __restrict__ entPart) {
  __shared__ float attn[ROWS_PER_BLK * 65];  // stride 65: conflict-free rows
  __shared__ bf16  wlds[ROWS_PER_BLK * 64];
  __shared__ float elds[ROWS_PER_BLK];

  const int tid = threadIdx.x;
  const int lane = tid & 31, wv = tid >> 5;
  const int m = lane & 15, half = lane >> 4;
  const int off = half * 8;
  const long rowbase = (long)blockIdx.x * ROWS_PER_BLK + wv * 32;  // wave: 32 rows
  const float* xrow0 = x + (rowbase + m) * DIM;
  const float* xrow1 = xrow0 + 16 * DIM;

  // ---- Stage 1: attn logits, 32 rows x 64 atoms per wave ----
  v8f p0 = {}, p1 = {}, p2 = {}, p3 = {};  // rows [0,16)  x atom tiles 0..3
  v8f q0 = {}, q1 = {}, q2 = {}, q3 = {};  // rows [16,32) x atom tiles 0..3
  const bf16* mrow = Mt + m * DIM + half * 16;
  for (int d0 = 0; d0 < DIM; d0 += 32) {
    v16bf A0, A1;
    const float* xp0 = xrow0 + d0 + half * 16;
    const float* xp1 = xrow1 + d0 + half * 16;
    __builtin_prefetch(xp0 + 256, 0, 0);   // stream-ahead hint (global_prefetch)
    __builtin_prefetch(xp1 + 256, 0, 0);
#pragma unroll
    for (int i = 0; i < 16; ++i) {
      A0[i] = (bf16)xp0[i];
      A1[i] = (bf16)xp1[i];
    }
    const bf16* bp = mrow + d0;
    v16bf B0 = *(const v16bf*)(bp);
    v16bf B1 = *(const v16bf*)(bp + 16 * DIM);
    v16bf B2 = *(const v16bf*)(bp + 32 * DIM);
    v16bf B3 = *(const v16bf*)(bp + 48 * DIM);
    p0 = wmma_bf16(A0, B0, p0);  q0 = wmma_bf16(A1, B0, q0);
    p1 = wmma_bf16(A0, B1, p1);  q1 = wmma_bf16(A1, B1, q1);
    p2 = wmma_bf16(A0, B2, p2);  q2 = wmma_bf16(A1, B2, q2);
    p3 = wmma_bf16(A0, B3, p3);  q3 = wmma_bf16(A1, B3, q3);
  }
  const float c0 = cvec[m], c1 = cvec[16 + m], c2 = cvec[32 + m], c3 = cvec[48 + m];
  const int rl0 = wv * 32 + 8 * half;
#pragma unroll
  for (int r = 0; r < 8; ++r) {
    float* ap = attn + (rl0 + r) * 65;
    ap[m]      = p0[r] + c0;
    ap[16 + m] = p1[r] + c1;
    ap[32 + m] = p2[r] + c2;
    ap[48 + m] = p3[r] + c3;
    float* aq = ap + 16 * 65;
    aq[m]      = q0[r] + c0;
    aq[16 + m] = q1[r] + c1;
    aq[32 + m] = q2[r] + c2;
    aq[48 + m] = q3[r] + c3;
  }
  __syncthreads();

  // ---- Stage 2: softmax + entropy, one thread per row (64 threads, 64 rows) ----
  {
    float* rp = attn + tid * 65;
    float mx = rp[0];
    for (int a = 1; a < NATOMS; ++a) mx = fmaxf(mx, rp[a]);
    float s = 0.f;
    for (int a = 0; a < NATOMS; ++a) { float e = __expf(rp[a] - mx); rp[a] = e; s += e; }
    const float inv = 1.f / s;
    float ent = 0.f;
    bf16* wp = wlds + tid * 64;
    for (int a = 0; a < NATOMS; ++a) {
      float w = rp[a] * inv;
      ent -= w * __logf(w + 1e-6f);
      wp[a] = (bf16)w;
    }
    elds[tid] = ent;
  }
  __syncthreads();
  if (tid == 0) {
    float s = 0.f;
    for (int r = 0; r < ROWS_PER_BLK; ++r) s += elds[r];
    entPart[blockIdx.x] = s;
  }

  // ---- Stage 3: x_clean = weights @ basis (2 row tiles share B operands) ----
  const bf16* wpa = wlds + (wv * 32 + m) * 64;
  const bf16* wpb = wpa + 16 * 64;
  v16bf W0a, W1a, W0b, W1b;
#pragma unroll
  for (int i = 0; i < 16; ++i) {
    W0a[i]     = wpa[half * 16 + i];   // K 0..31
    W1a[i]     = wpa[32 + half * 16 + i];
    W0b[i]     = wpb[half * 16 + i];
    W1b[i]     = wpb[32 + half * 16 + i];
  }
  float* orow0 = out + (rowbase + 8 * half) * DIM + m;
  float* orow1 = orow0 + 16 * DIM;
  for (int dt = 0; dt < 64; ++dt) {
    const bf16* bp = basisT + (dt * 16 + m) * 64 + half * 16;
    v16bf B0 = *(const v16bf*)(bp);
    v16bf B1 = *(const v16bf*)(bp + 32);
    v8f acca = {}, accb = {};
    acca = wmma_bf16(W0a, B0, acca);
    acca = wmma_bf16(W1a, B1, acca);
    accb = wmma_bf16(W0b, B0, accb);
    accb = wmma_bf16(W1b, B1, accb);
    float* opa = orow0 + dt * 16;
    float* opb = orow1 + dt * 16;
#pragma unroll
    for (int r = 0; r < 8; ++r) {
      opa[(long)r * DIM] = acca[r];
      opb[(long)r * DIM] = accb[r];
    }
  }
}

// ---------------------------------------------------------------------------
// Final: deterministic reduce of per-block entropy partials -> mean
// ---------------------------------------------------------------------------
__global__ __launch_bounds__(NBLK) void kernel_final(const float* __restrict__ part,
                                                     float* __restrict__ dst) {
  __shared__ float red[NBLK];
  const int t = threadIdx.x;
  red[t] = part[t];
  __syncthreads();
  for (int st = NBLK / 2; st > 0; st >>= 1) {
    if (t < st) red[t] += red[t + st];
    __syncthreads();
  }
  if (t == 0) dst[0] = red[0] * (1.0f / (float)NROWS);
}

// ---------------------------------------------------------------------------
extern "C" void kernel_launch(void* const* d_in, const int* in_sizes, int n_in,
                              void* d_out, int out_size, void* d_ws, size_t ws_size,
                              hipStream_t stream) {
  (void)in_sizes; (void)n_in; (void)ws_size;
  const float* x     = (const float*)d_in[0];
  const float* basis = (const float*)d_in[1];
  const float* Wq    = (const float*)d_in[2];
  const float* bq    = (const float*)d_in[3];
  const float* Wk    = (const float*)d_in[4];
  const float* bk    = (const float*)d_in[5];
  float* out = (float*)d_out;

  char* ws = (char*)d_ws;
  float* Kmat   = (float*)(ws + 0);                  // 64*1024*4   = 262144 B
  bf16*  Mt     = (bf16*) (ws + 262144);             // 64*1024*2   = 131072 B
  bf16*  basisT = (bf16*) (ws + 262144 + 131072);    // 1024*64*2   = 131072 B
  float* cvec   = (float*)(ws + 524288);             // 64*4
  float* entp   = (float*)(ws + 524544);             // NBLK*4

  const float scale = 0.03125f;  // 1024^-0.5

  kernel_K    <<<dim3(4, 64), 32, 0, stream>>>(basis, Wk, bk, Kmat);
  kernel_Mt   <<<64, 256, 0, stream>>>(Kmat, Wq, Mt, scale);
  kernel_c    <<<64, 256, 0, stream>>>(Kmat, bq, cvec, scale);
  kernel_bt   <<<256, 256, 0, stream>>>(basis, basisT);
  kernel_main <<<NBLK, 64, 0, stream>>>(x, Mt, cvec, basisT, out, entp);
  kernel_final<<<1, NBLK, 0, stream>>>(entp, out + (out_size - 1));
}
